// DeformAttnRoIPool_90331752169714
// MI455X (gfx1250) — compile-verified, matrix operand order checked by reading, not run on hardware
//
#include <hip/hip_runtime.h>
#include <hip/hip_bf16.h>

// ---------------- problem constants ----------------
#define FB   4
#define FC   256
#define FH   128
#define FW   128
#define NROI 512
#define OH   7
#define OW   7
#define NP   4
#define M1   (NROI * OH * OW)   // 25088 rows per p
#define MTOT (NP * M1)          // 100352 rows total
#define CD   256                // channel / head dim

typedef __attribute__((ext_vector_type(16))) __bf16 v16bf;
typedef __attribute__((ext_vector_type(8)))  __bf16 v8bf;
typedef __attribute__((ext_vector_type(8)))  float  v8f;

// async-to-LDS builtin operand types: v4i in addrspace(1)/(3)
typedef int vs4i __attribute__((__vector_size__(4 * sizeof(int))));
typedef __attribute__((address_space(1))) vs4i as1_v4i;
typedef __attribute__((address_space(3))) vs4i as3_v4i;

// ---- CDNA5 async global->LDS copy (16B), ASYNCcnt-tracked ----
__device__ __forceinline__ void async_copy_b128(const void* gsrc, void* ldst) {
  as1_v4i* gp = (as1_v4i*)(unsigned long long)(uintptr_t)gsrc;
  as3_v4i* lp = (as3_v4i*)(unsigned int)(uintptr_t)ldst;   // low 32 bits == LDS offset
#if __has_builtin(__builtin_amdgcn_global_load_async_to_lds_b128)
  __builtin_amdgcn_global_load_async_to_lds_b128(gp, lp, 0, 0);
#else
  asm volatile("global_load_async_to_lds_b128 %0, %1, off"
               :: "v"((unsigned int)(uintptr_t)ldst),
                  "v"((unsigned long long)(uintptr_t)gsrc) : "memory");
#endif
}

__device__ __forceinline__ void wait_asynccnt_le8() {
#if __has_builtin(__builtin_amdgcn_s_wait_asynccnt)
  __builtin_amdgcn_s_wait_asynccnt(8);
#else
  asm volatile("s_wait_asynccnt 0x8" ::: "memory");
#endif
}
__device__ __forceinline__ void wait_asynccnt_0() {
#if __has_builtin(__builtin_amdgcn_s_wait_asynccnt)
  __builtin_amdgcn_s_wait_asynccnt(0);
#else
  asm volatile("s_wait_asynccnt 0x0" ::: "memory");
#endif
}

// Load a 16x32 bf16 WMMA fragment from a row-major (ld elements) matrix.
// ISA 7.12.2: lanes 0-15 hold rows M=0..15 with K octets {0..7,16..23},
// lanes 16-31 hold the same rows with K octets {8..15,24..31}.
// Weights are stored N-major (NxK), so B fragments use the same loader.
__device__ __forceinline__ v16bf load_frag(const __bf16* base, int ld) {
  const int lane = threadIdx.x & 31;
  const __bf16* p = base + (lane & 15) * ld + ((lane >> 4) << 3);
  v8bf lo = *(const v8bf*)p;         // K = kh .. kh+7
  v8bf hi = *(const v8bf*)(p + 16);  // K = 16+kh .. 16+kh+7
  return __builtin_shufflevector(lo, hi, 0,1,2,3,4,5,6,7,8,9,10,11,12,13,14,15);
}

// ------------- kernel 1: fused weight prep -------------
// blocks 0..255   : WkP[d][c] = sum_j Wk[d][j]*wproj[j][c]   (+ bias fold)
// blocks 256..511 : WvP likewise
// blocks 512..767 : WoutB = bf16(out_proj_w)
// blocks 768..816 : Qs[i][d] = queries[i] . Wq[d] + bq[d]   (49 distinct Q rows)
__global__ __launch_bounds__(256) void prep_weights_kernel(
    const float* __restrict__ w_proj, const float* __restrict__ b_proj,
    const float* __restrict__ in_proj_w, const float* __restrict__ in_proj_b,
    const float* __restrict__ out_proj_w, const float* __restrict__ queries,
    __bf16* __restrict__ WkP, __bf16* __restrict__ WvP, __bf16* __restrict__ WoutB,
    float* __restrict__ bkP, float* __restrict__ bvP, float* __restrict__ Qs)
{
  const int bid = blockIdx.x;
  const int t   = threadIdx.x;
  if (bid < 512) {
    const int d      = bid & 255;
    const int sel    = bid >> 8;                       // 0 -> K, 1 -> V
    const float* Wrow = in_proj_w + (size_t)(256 + sel * 256 + d) * 256;
    float acc = 0.f;
    for (int j = 0; j < 256; ++j) acc += Wrow[j] * w_proj[j * 256 + t];
    (sel == 0 ? WkP : WvP)[d * 256 + t] = (__bf16)acc;
    if (t == 0) {
      float bb = in_proj_b[256 + sel * 256 + d];
      for (int j = 0; j < 256; ++j) bb += Wrow[j] * b_proj[j];
      (sel == 0 ? bkP : bvP)[d] = bb;
    }
  } else if (bid < 768) {
    const int d = bid - 512;
    WoutB[d * 256 + t] = (__bf16)out_proj_w[d * 256 + t];
  } else {
    const int i = bid - 768;                           // 0..48
    const float* qrow = queries + i * 256;
    const float* wq   = in_proj_w + (size_t)t * 256;   // Wq row t
    float acc = in_proj_b[t];
    for (int c = 0; c < 256; ++c) acc += qrow[c] * wq[c];
    Qs[i * 256 + t] = acc;
  }
}

// ------------- kernel 2: (B,C,H,W) f32 -> (B,HW,C) bf16 transpose -------------
__global__ __launch_bounds__(256) void transpose_feat_kernel(
    const float* __restrict__ f, __bf16* __restrict__ ft)
{
  __shared__ float tile[32][33];
  const int b  = blockIdx.z;
  const int c0 = blockIdx.y * 32;
  const int p0 = blockIdx.x * 32;
  const int tx = threadIdx.x & 31;
  const int ty = threadIdx.x >> 5;
  const float* src = f + ((size_t)b * 256 + c0) * (FH * FW) + p0;
  #pragma unroll
  for (int i = 0; i < 32; i += 8)
    tile[ty + i][tx] = src[(size_t)(ty + i) * (FH * FW) + tx];
  __syncthreads();
  __bf16* dst = ft + ((size_t)b * (FH * FW) + p0) * 256 + c0;
  #pragma unroll
  for (int i = 0; i < 32; i += 8)
    dst[(size_t)(ty + i) * 256 + tx] = (__bf16)tile[tx][ty + i];
}

// ------------- kernel 3: deformable RoI bilinear sampling -------------
// One wave per (p, roi, bin). Lanes cover 256 channels (8 each, 16B vectors).
__global__ __launch_bounds__(256) void sample_kernel(
    const __bf16* __restrict__ featT, const float* __restrict__ rois,
    const float* __restrict__ offs, __bf16* __restrict__ S)
{
  const int gw   = blockIdx.x * 8 + (threadIdx.x >> 5);   // exact: 12544*8 == MTOT
  const int lane = threadIdx.x & 31;
  const int p  = gw / M1;
  const int rr = gw - p * M1;
  const int n  = rr / (OH * OW);
  const int hw = rr - n * (OH * OW);
  const int h  = hw / OW, w = hw - h * OW;

  const float* r = rois + n * 5;
  const int   b  = (int)r[0];
  const float x1 = r[1] * 0.0625f - 0.5f;
  const float y1 = r[2] * 0.0625f - 0.5f;
  const float rw = r[3] * 0.0625f - 0.5f - x1;
  const float rh = r[4] * 0.0625f - 0.5f - y1;
  const float* off = offs + (((size_t)p * OH + h) * OW + w) * 2;
  const float x = x1 + (w + 0.5f) * (rw * (1.f / OW)) + 0.1f * rw * off[0];
  const float y = y1 + (h + 0.5f) * (rh * (1.f / OH)) + 0.1f * rh * off[1];
  const bool valid = (y > -1.f) && (y < (float)FH) && (x > -1.f) && (x < (float)FW);
  const float yc = fminf(fmaxf(y, 0.f), (float)(FH - 1));
  const float xc = fminf(fmaxf(x, 0.f), (float)(FW - 1));
  const int y0 = (int)floorf(yc), x0 = (int)floorf(xc);
  const int y1i = (y0 + 1 < FH) ? y0 + 1 : FH - 1;
  const int x1i = (x0 + 1 < FW) ? x0 + 1 : FW - 1;
  const float ly = yc - (float)y0, lx = xc - (float)x0;
  const float hy = 1.f - ly, hx = 1.f - lx;
  const float w00 = hy * hx, w01 = hy * lx, w10 = ly * hx, w11 = ly * lx;

  const size_t bbase = (size_t)b * (FH * FW) * 256 + (size_t)lane * 8;
  const v8bf c00 = *(const v8bf*)(featT + bbase + (size_t)(y0  * FW + x0 ) * 256);
  const v8bf c01 = *(const v8bf*)(featT + bbase + (size_t)(y0  * FW + x1i) * 256);
  const v8bf c10 = *(const v8bf*)(featT + bbase + (size_t)(y1i * FW + x0 ) * 256);
  const v8bf c11 = *(const v8bf*)(featT + bbase + (size_t)(y1i * FW + x1i) * 256);

  v8bf res;
  #pragma unroll
  for (int i = 0; i < 8; ++i) {
    float v = w00 * (float)c00[i] + w01 * (float)c01[i] +
              w10 * (float)c10[i] + w11 * (float)c11[i];
    res[i] = (__bf16)(valid ? v : 0.f);
  }
  *(v8bf*)(S + (size_t)gw * 256 + (size_t)lane * 8) = res;
}

// ------------- kernel 4/6: bf16 WMMA GEMM with double-buffered async LDS staging ----
//   Out[M][256] = A[M][256] @ W[N=256][K=256]^T + bias
// 8 waves/block, 16 rows/wave (128 rows/block). Weight K-chunks (256x64 bf16 = 32KB)
// stream into LDS via GLOBAL_LOAD_ASYNC_TO_LDS_B128 (ASYNCcnt), double-buffered so
// chunk kc+64 is in flight while WMMAs consume chunk kc.
// SCATTER=false: bf16 row-major output.  SCATTER=true: f32 scatter to (N, C, 7, 7).
template <bool SCATTER>
__global__ __launch_bounds__(256) void gemm_wmma_kernel(
    const __bf16* __restrict__ Amat, const __bf16* __restrict__ Wmat,
    const float* __restrict__ bias, void* __restrict__ OutP)
{
  __shared__ __align__(16) __bf16 ldsW[2][256 * 64];  // 2 x [N=256][Kchunk=64]
  const int wave = threadIdx.x >> 5;
  const int lane = threadIdx.x & 31;
  const int m0   = blockIdx.x * 128 + wave * 16;

  // stage K-chunk kc into buffer buf: thread t copies W row t (64 bf16 = 8 x b128)
  auto stage = [&](int buf, int kc) {
    const __bf16* src = Wmat + (size_t)threadIdx.x * 256 + kc;
    __bf16* dst = &ldsW[buf][(size_t)threadIdx.x * 64];
    #pragma unroll
    for (int i = 0; i < 8; ++i)
      async_copy_b128(src + i * 8, dst + i * 8);
  };

  v8f acc[16] = {};                                   // 16x256 f32 accumulators
  stage(0, 0);                                        // prime the pipeline (8 in flight)
  #pragma unroll
  for (int it = 0; it < 4; ++it) {
    const int kc = it * 64;
    if (it < 3) {
      stage((it + 1) & 1, kc + 64);                   // 16 in flight
      wait_asynccnt_le8();                            // in-order: current chunk landed
    } else {
      wait_asynccnt_0();
    }
    __syncthreads();                                  // all waves see the chunk
    __builtin_prefetch(Amat + (size_t)m0 * 256 + kc + 64, 0, 1);
    const __bf16* wbuf = ldsW[it & 1];
    #pragma unroll
    for (int ks = 0; ks < 2; ++ks) {
      const int k0 = kc + ks * 32;
      const v16bf afrag = load_frag(Amat + (size_t)m0 * 256 + k0, 256);
      #pragma unroll
      for (int nb = 0; nb < 16; ++nb) {
        const v16bf bfrag = load_frag(wbuf + nb * 16 * 64 + ks * 32, 64);
        acc[nb] = __builtin_amdgcn_wmma_f32_16x16x32_bf16(
            false, afrag, false, bfrag, (short)0, acc[nb], false, false);
      }
    }
    __syncthreads();                                  // done reading before next overwrite
  }

  // epilogue: C layout — VGPR j holds rows m0+j (lanes 0-15) / m0+8+j (lanes 16-31)
  const int rl  = (lane >> 4) * 8;
  const int col = lane & 15;
  if (SCATTER) {
    float* out = (float*)OutP;                        // (N, 256, 7, 7) f32
    #pragma unroll
    for (int nb = 0; nb < 16; ++nb) {
      const int d  = nb * 16 + col;
      const float bb = bias[d];
      #pragma unroll
      for (int j = 0; j < 8; ++j) {
        const int row = m0 + rl + j;                  // row = n*49 + hw
        const int n   = row / (OH * OW);
        const int hw  = row - n * (OH * OW);
        out[((size_t)n * 256 + d) * (OH * OW) + hw] = acc[nb][j] + bb;
      }
    }
  } else {
    __bf16* out = (__bf16*)OutP;                      // [M][256] bf16
    #pragma unroll
    for (int nb = 0; nb < 16; ++nb) {
      const int nn = nb * 16 + col;
      const float bb = bias[nn];
      #pragma unroll
      for (int j = 0; j < 8; ++j)
        out[(size_t)(m0 + rl + j) * 256 + nn] = (__bf16)(acc[nb][j] + bb);
    }
  }
}

// ------------- kernel 5: attention over P=4 + V blend -------------
__global__ __launch_bounds__(256) void attn_kernel(
    const float* __restrict__ Qs, const __bf16* __restrict__ Kmat,
    const __bf16* __restrict__ Vmat, __bf16* __restrict__ Aout)
{
  const int gw   = blockIdx.x * 8 + (threadIdx.x >> 5);  // exact: 3136*8 == M1
  const int lane = threadIdx.x & 31;
  const int hw   = gw % (OH * OW);

  float qv[8];
  const float* q = Qs + (size_t)hw * 256 + (size_t)lane * 8;
  #pragma unroll
  for (int i = 0; i < 8; ++i) qv[i] = q[i];

  float s[NP];
  #pragma unroll
  for (int p = 0; p < NP; ++p) {
    const v8bf k = *(const v8bf*)(Kmat + ((size_t)p * M1 + gw) * 256 + (size_t)lane * 8);
    float d = 0.f;
    #pragma unroll
    for (int i = 0; i < 8; ++i) d += qv[i] * (float)k[i];
    #pragma unroll
    for (int o = 16; o > 0; o >>= 1) d += __shfl_xor(d, o, 32);
    s[p] = d * 0.0625f;                                  // * d^-0.5 = 1/16
  }
  float m = s[0];
  #pragma unroll
  for (int p = 1; p < NP; ++p) m = fmaxf(m, s[p]);
  float e[NP], sum = 0.f;
  #pragma unroll
  for (int p = 0; p < NP; ++p) { e[p] = __expf(s[p] - m); sum += e[p]; }
  const float inv = 1.f / sum;

  float accv[8] = {0.f, 0.f, 0.f, 0.f, 0.f, 0.f, 0.f, 0.f};
  #pragma unroll
  for (int p = 0; p < NP; ++p) {
    const float a = e[p] * inv;
    const v8bf v = *(const v8bf*)(Vmat + ((size_t)p * M1 + gw) * 256 + (size_t)lane * 8);
    #pragma unroll
    for (int i = 0; i < 8; ++i) accv[i] += a * (float)v[i];
  }
  v8bf outv;
  #pragma unroll
  for (int i = 0; i < 8; ++i) outv[i] = (__bf16)accv[i];
  *(v8bf*)(Aout + (size_t)gw * 256 + (size_t)lane * 8) = outv;
}

// ---------------- host-side launcher ----------------
extern "C" void kernel_launch(void* const* d_in, const int* in_sizes, int n_in,
                              void* d_out, int out_size, void* d_ws, size_t ws_size,
                              hipStream_t stream) {
  (void)in_sizes; (void)n_in; (void)out_size; (void)ws_size;
  const float* features   = (const float*)d_in[0];
  const float* rois       = (const float*)d_in[1];
  const float* ppo        = (const float*)d_in[2];
  const float* queries    = (const float*)d_in[3];
  const float* w_proj     = (const float*)d_in[4];
  const float* b_proj     = (const float*)d_in[5];
  const float* in_proj_w  = (const float*)d_in[6];
  const float* in_proj_b  = (const float*)d_in[7];
  const float* out_proj_w = (const float*)d_in[8];
  const float* out_proj_b = (const float*)d_in[9];
  float* out = (float*)d_out;

  char* ws = (char*)d_ws;
  size_t off = 0;
  auto take = [&](size_t bytes) { char* p = ws + off; off += (bytes + 255) & ~(size_t)255; return p; };
  __bf16* featT = (__bf16*)take((size_t)FB * FH * FW * 256 * 2);   // 32 MB
  __bf16* S     = (__bf16*)take((size_t)MTOT * 256 * 2);           // 51 MB
  __bf16* Kmat  = (__bf16*)take((size_t)MTOT * 256 * 2);           // 51 MB
  __bf16* Vmat  = (__bf16*)take((size_t)MTOT * 256 * 2);           // 51 MB
  __bf16* Aout  = (__bf16*)take((size_t)M1 * 256 * 2);             // 13 MB
  __bf16* WkP   = (__bf16*)take(256 * 256 * 2);
  __bf16* WvP   = (__bf16*)take(256 * 256 * 2);
  __bf16* WoutB = (__bf16*)take(256 * 256 * 2);
  float*  bkP   = (float*)take(256 * 4);
  float*  bvP   = (float*)take(256 * 4);
  float*  Qs    = (float*)take(49 * 256 * 4);

  prep_weights_kernel<<<dim3(817), dim3(256), 0, stream>>>(
      w_proj, b_proj, in_proj_w, in_proj_b, out_proj_w, queries,
      WkP, WvP, WoutB, bkP, bvP, Qs);

  transpose_feat_kernel<<<dim3((FH * FW) / 32, 256 / 32, FB), dim3(256), 0, stream>>>(
      features, featT);

  sample_kernel<<<dim3(MTOT / 8), dim3(256), 0, stream>>>(featT, rois, ppo, S);

  gemm_wmma_kernel<false><<<dim3(MTOT / 128), dim3(256), 0, stream>>>(S, WkP, bkP, (void*)Kmat);
  gemm_wmma_kernel<false><<<dim3(MTOT / 128), dim3(256), 0, stream>>>(S, WvP, bvP, (void*)Vmat);

  attn_kernel<<<dim3(M1 / 8), dim3(256), 0, stream>>>(Qs, Kmat, Vmat, Aout);

  gemm_wmma_kernel<true><<<dim3(M1 / 128), dim3(256), 0, stream>>>(Aout, WoutB, out_proj_b, (void*)out);
}
